// GatingNetwork_14516989460789
// MI455X (gfx1250) — matrix operand order repaired and checked by reading, not compiled, not run on hardware
//
#include <hip/hip_runtime.h>
#include <cstdint>

// Problem constants
#define B_  8192
#define D_  1024
#define H_  4096
#define O_  1024
#define E_  8

// Tiling
#define BM    64            // routed tokens per block
#define NTHR  512           // 16 wave32s
#define HC    256           // H chunk held in LDS
#define KC1   64            // K chunk for GEMM1 staging
#define KC2   32            // K chunk for GEMM2 staging

// LDS pitches (ushort units). Chosen for alignment + bank spread:
//  PX, PH: 16B-aligned rows (b128 A-frag reads), pitch/4 dwords == 4 (mod 32)
//  PW1, PW2: 8B-aligned rows (b64 B-frag reads), 2-way worst-case conflicts
#define PX  (KC1 + 8)       // 72
#define PW1 (KC1 + 4)       // 68
#define PW2 (KC2 + 4)       // 36
#define PH  (HC  + 8)       // 264

typedef unsigned short u16;
typedef float        v4f  __attribute__((ext_vector_type(4)));
typedef float        v8f  __attribute__((ext_vector_type(8)));
typedef unsigned int v2u  __attribute__((ext_vector_type(2)));
typedef unsigned int v4u  __attribute__((ext_vector_type(4)));
typedef __bf16       v2bf  __attribute__((ext_vector_type(2)));
typedef __bf16       v16bf __attribute__((ext_vector_type(16)));

union FragA { v4u q[2]; v16bf v; };   // 16x32 bf16 A fragment (32B/lane)
union FragB { v2u d[4]; v16bf v; };   // 32x16 bf16 B fragment (32B/lane)

__device__ __forceinline__ v8f wmma_bf16(v16bf a, v16bf b, v8f c) {
#if defined(__HIP_DEVICE_COMPILE__)
    return __builtin_amdgcn_wmma_f32_16x16x32_bf16(false, a, false, b, (short)0, c,
                                                   false, false);
#else
    (void)a; (void)b;
    return c;
#endif
}

__device__ __forceinline__ u16 f2bf(float f) {          // RNE float -> bf16
    unsigned u = __builtin_bit_cast(unsigned, f);
    return (u16)((u + 0x7fffu + ((u >> 16) & 1u)) >> 16);
}

// pack two floats to packed bf16 (round-half-up): 3 VALU per 2 elements
__device__ __forceinline__ unsigned pack2(float a, float b) {
#if defined(__HIP_DEVICE_COMPILE__) && __has_builtin(__builtin_amdgcn_cvt_pk_bf16_f32)
    v2bf r = __builtin_amdgcn_cvt_pk_bf16_f32(a, b);
    return __builtin_bit_cast(unsigned, r);
#elif defined(__HIP_DEVICE_COMPILE__) && __has_builtin(__builtin_amdgcn_perm)
    unsigned ua = __builtin_bit_cast(unsigned, a) + 0x8000u;
    unsigned ub = __builtin_bit_cast(unsigned, b) + 0x8000u;
    return __builtin_amdgcn_perm(ub, ua, 0x07060302u);  // {ub[31:16], ua[31:16]}
#else
    return (unsigned)f2bf(a) | ((unsigned)f2bf(b) << 16);
#endif
}

// ---------------- routing: per-expert token lists (duplicates merged) -------
__global__ void route_kernel(const int* __restrict__ topk,
                             const float* __restrict__ wgt,
                             int* __restrict__ cnt,
                             int* __restrict__ toks,
                             float* __restrict__ gates) {
    int b = blockIdx.x * blockDim.x + threadIdx.x;
    if (b >= B_) return;
    int   e0 = topk[b * 2 + 0], e1 = topk[b * 2 + 1];
    float w0 = wgt[b * 2 + 0],  w1 = wgt[b * 2 + 1];
    if (e0 == e1) {
        int s = atomicAdd(&cnt[e0], 1);
        toks[e0 * B_ + s] = b;  gates[e0 * B_ + s] = w0 + w1;
    } else {
        int s0 = atomicAdd(&cnt[e0], 1);
        toks[e0 * B_ + s0] = b; gates[e0 * B_ + s0] = w0;
        int s1 = atomicAdd(&cnt[e1], 1);
        toks[e1 * B_ + s1] = b; gates[e1 * B_ + s1] = w1;
    }
}

// ---------------- fused expert MLP over routed tokens -----------------------
__global__ __launch_bounds__(NTHR, 1)
void moe_gemm(const float* __restrict__ x,
              const float* __restrict__ W1, const float* __restrict__ b1,
              const float* __restrict__ W2, const float* __restrict__ b2,
              float* __restrict__ out,
              const int* __restrict__ cnt,
              const int* __restrict__ toks,
              const float* __restrict__ gates)
{
    __shared__ __align__(16) u16 s_x[BM * PX];     //  9.2 KB x chunk (bf16)
    __shared__ __align__(16) u16 s_w[O_ * PW2];    // 72   KB W chunk, transposed
    __shared__ __align__(16) u16 s_h[BM * PH];     // 33.8 KB activations (bf16)
    __shared__ int   s_tok[BM];
    __shared__ float s_gate[BM];

    const int e     = blockIdx.y;
    const int count = cnt[e];
    const int t0    = blockIdx.x * BM;
    if (t0 >= count) return;
    const int MT = (count - t0 < BM) ? (count - t0) : BM;

    const int tid  = threadIdx.x;
    const int lane = tid & 31;
    const int wave = tid >> 5;        // 0..15
    const int lo   = lane & 15;
    const int hi   = lane >> 4;
    const int mq   = wave & 3;        // M tile (16 rows each)
    const int ng   = wave >> 2;       // N group

    if (tid < BM) {
        int idx = (tid < MT) ? (t0 + tid) : t0;   // pad rows alias row 0 (gate 0)
        s_tok[tid]  = toks[e * B_ + idx];
        s_gate[tid] = (tid < MT) ? gates[e * B_ + idx] : 0.f;
    }
    __syncthreads();

    const float* W1e = W1 + (size_t)e * D_ * H_;
    const float* b1e = b1 + (size_t)e * H_;
    const float* W2e = W2 + (size_t)e * H_ * O_;
    const float* b2e = b2 + (size_t)e * O_;

    // staging coordinates
    const int xm = tid >> 3, xc = (tid & 7) * 8;              // x chunk
    const long xrowbase = (long)s_tok[xm] * D_ + xc;
    const int w1k = (tid & 15) * 4, w1n = (tid >> 4) * 4;     // W1 4x4 blocks
    const int w2k = (tid & 7)  * 4, w2n = (tid >> 3) * 4;     // W2 4x4 blocks

    // fragment base pointers (constant per thread)
    const u16* axp = s_x + (mq * 16 + lo) * PX + 16 * hi;
    const u16* ahp = s_h + (mq * 16 + lo) * PH + 16 * hi;
    const u16* bw1 = s_w + lo * PW1 + 16 * hi;
    const u16* bw2 = s_w + (ng * 256 + lo) * PW2 + 16 * hi;

    v8f oacc[16];
    #pragma unroll
    for (int j = 0; j < 16; ++j)
        #pragma unroll
        for (int i = 0; i < 8; ++i) oacc[j][i] = 0.f;

    for (int hc = 0; hc < H_ / HC; ++hc) {
        const int h0 = hc * HC;

        // ---- GEMM1: hacc = x @ W1[:, h0:h0+HC] + b1 ----
        v8f hacc[4];
        #pragma unroll
        for (int j = 0; j < 4; ++j) {
            float bb = b1e[h0 + (ng * 4 + j) * 16 + lo];
            #pragma unroll
            for (int i = 0; i < 8; ++i) hacc[j][i] = bb;
        }

        for (int kc = 0; kc < D_ / KC1; ++kc) {
            __syncthreads();
            { // stage x chunk (row-major bf16)
                v4f f0 = *(const v4f*)(x + xrowbase + kc * KC1);
                v4f f1 = *(const v4f*)(x + xrowbase + kc * KC1 + 4);
                v4u pk = { pack2(f0[0], f0[1]), pack2(f0[2], f0[3]),
                           pack2(f1[0], f1[1]), pack2(f1[2], f1[3]) };
                *(v4u*)(s_x + xm * PX + xc) = pk;
            }
            // stage W1 chunk transposed: s_w[n][k]
            #pragma unroll
            for (int i = 0; i < 2; ++i) {
                const int n = w1n + i * 128;
                v4f blk[4];
                #pragma unroll
                for (int c = 0; c < 4; ++c)
                    blk[c] = *(const v4f*)(W1e + (size_t)(kc * KC1 + w1k + c) * H_ + h0 + n);
                #pragma unroll
                for (int c2 = 0; c2 < 4; ++c2) {
                    v2u pkt = { pack2(blk[0][c2], blk[1][c2]),
                                pack2(blk[2][c2], blk[3][c2]) };
                    *(v2u*)(s_w + (n + c2) * PW1 + w1k) = pkt;
                }
            }
            __syncthreads();

            // compute: depth-2 software pipeline on B fragments
            FragA a0, a1;
            a0.q[0] = *(const v4u*)axp;        a0.q[1] = *(const v4u*)(axp + 8);
            a1.q[0] = *(const v4u*)(axp + 32); a1.q[1] = *(const v4u*)(axp + 40);
            FragB cur, nxt;
            #pragma unroll
            for (int p = 0; p < 2; ++p) {
                const u16* bp = bw1 + ((ng * 4 + p) * 16) * PW1;
                FragB t;
                t.d[0] = *(const v2u*)bp;      t.d[1] = *(const v2u*)(bp + 4);
                t.d[2] = *(const v2u*)(bp + 8); t.d[3] = *(const v2u*)(bp + 12);
                if (p == 0) cur = t; else nxt = t;
            }
            #pragma unroll
            for (int u = 0; u < 8; ++u) {      // u = kk*4 + j
                FragB nn = nxt;
                if (u + 2 < 8) {
                    const int un = u + 2;
                    const u16* bp = bw1 + ((ng * 4 + (un & 3)) * 16) * PW1 + (un >> 2) * 32;
                    nn.d[0] = *(const v2u*)bp;      nn.d[1] = *(const v2u*)(bp + 4);
                    nn.d[2] = *(const v2u*)(bp + 8); nn.d[3] = *(const v2u*)(bp + 12);
                }
                hacc[u & 3] = wmma_bf16((u < 4) ? a0.v : a1.v, cur.v, hacc[u & 3]);
                cur = nxt; nxt = nn;
            }
        }

        // ---- relu -> bf16 activations in LDS (round-half-up, d16_hi store) --
        __syncthreads();   // previous phase-2 readers of s_h are done
        #pragma unroll
        for (int j = 0; j < 4; ++j) {
            const int coll = (ng * 4 + j) * 16 + lo;
            #pragma unroll
            for (int i = 0; i < 8; ++i) {
                const int row = mq * 16 + 8 * hi + i;
                unsigned u = __builtin_bit_cast(unsigned, fmaxf(hacc[j][i], 0.f)) + 0x8000u;
                s_h[row * PH + coll] = (u16)(u >> 16);
            }
        }

        // ---- GEMM2: oacc += relu(h) @ W2[h0:h0+HC, :] ----
        for (int kc2 = 0; kc2 < HC / KC2; ++kc2) {
            __syncthreads();   // also publishes s_h on first iteration
            // stage W2 chunk transposed: s_w[n][k]
            #pragma unroll
            for (int i = 0; i < 4; ++i) {
                const int n = w2n + i * 256;
                v4f blk[4];
                #pragma unroll
                for (int c = 0; c < 4; ++c)
                    blk[c] = *(const v4f*)(W2e + (size_t)(h0 + kc2 * KC2 + w2k + c) * O_ + n);
                #pragma unroll
                for (int c2 = 0; c2 < 4; ++c2) {
                    v2u pkt = { pack2(blk[0][c2], blk[1][c2]),
                                pack2(blk[2][c2], blk[3][c2]) };
                    *(v2u*)(s_w + (n + c2) * PW2 + w2k) = pkt;
                }
            }
            __syncthreads();

            FragA a;
            const u16* ap = ahp + kc2 * KC2;
            a.q[0] = *(const v4u*)ap;
            a.q[1] = *(const v4u*)(ap + 8);
            FragB cur, nxt;
            #pragma unroll
            for (int p = 0; p < 2; ++p) {
                const u16* bp = bw2 + (p * 16) * PW2;
                FragB t;
                t.d[0] = *(const v2u*)bp;      t.d[1] = *(const v2u*)(bp + 4);
                t.d[2] = *(const v2u*)(bp + 8); t.d[3] = *(const v2u*)(bp + 12);
                if (p == 0) cur = t; else nxt = t;
            }
            #pragma unroll
            for (int j = 0; j < 16; ++j) {
                FragB nn = nxt;
                if (j + 2 < 16) {
                    const u16* bp = bw2 + ((j + 2) * 16) * PW2;
                    nn.d[0] = *(const v2u*)bp;      nn.d[1] = *(const v2u*)(bp + 4);
                    nn.d[2] = *(const v2u*)(bp + 8); nn.d[3] = *(const v2u*)(bp + 12);
                }
                oacc[j] = wmma_bf16(a.v, cur.v, oacc[j]);
                cur = nxt; nxt = nn;
            }
        }
    }

    // ---- writeback: out[tok] += gate * (acc + b2). <=2 adds per element
    //      (K=2, duplicates merged) -> commutative -> deterministic.
    int   tokr[8]; float gr[8]; bool vr[8];
    #pragma unroll
    for (int i = 0; i < 8; ++i) {
        int row = mq * 16 + 8 * hi + i;
        tokr[i] = s_tok[row]; gr[i] = s_gate[row]; vr[i] = row < MT;
    }
    #pragma unroll
    for (int j = 0; j < 16; ++j) {
        const int col = ng * 256 + j * 16 + lo;
        const float bb = b2e[col];
        #pragma unroll
        for (int i = 0; i < 8; ++i) {
            if (vr[i])
                atomicAdd(out + (size_t)tokr[i] * O_ + col, gr[i] * (oacc[j][i] + bb));
        }
    }
}

extern "C" void kernel_launch(void* const* d_in, const int* in_sizes, int n_in,
                              void* d_out, int out_size, void* d_ws, size_t ws_size,
                              hipStream_t stream) {
    (void)in_sizes; (void)n_in; (void)out_size; (void)ws_size;
    const float* x    = (const float*)d_in[0];
    const int*   topk = (const int*)  d_in[1];
    const float* w    = (const float*)d_in[2];
    const float* W1   = (const float*)d_in[3];
    const float* b1   = (const float*)d_in[4];
    const float* W2   = (const float*)d_in[5];
    const float* b2   = (const float*)d_in[6];
    float*       out  = (float*)d_out;

    // workspace layout (~513 KB): counts | token lists | gate lists
    int*   cnt   = (int*)d_ws;
    int*   toks  = (int*)((char*)d_ws + 1024);
    float* gates = (float*)((char*)d_ws + 1024 + (size_t)E_ * B_ * 4);

    hipMemsetAsync(cnt, 0, 1024, stream);
    hipMemsetAsync(d_out, 0, (size_t)B_ * O_ * sizeof(float), stream);

    route_kernel<<<B_ / 256, 256, 0, stream>>>(topk, w, cnt, toks, gates);

    // One fat launch: (token-chunk, expert). Empty chunks exit immediately.
    moe_gemm<<<dim3(B_ / BM, E_), NTHR, 0, stream>>>(x, W1, b1, W2, b2, out,
                                                     cnt, toks, gates);
}
